// Block_wCB_54984171323396
// MI455X (gfx1250) — matrix operand order repaired and checked by reading, not compiled, not run on hardware
//
#include <hip/hip_runtime.h>

// CDNA5 / gfx1250: wave32, WMMA bf16 16x16x32 with f32 accumulation.
typedef __attribute__((ext_vector_type(16))) __bf16 v16bf;
typedef __attribute__((ext_vector_type(8)))  float  v8f;

#define N8K   8192
#define FDIM  64
#define BATCH 4
#define BNF   (N8K * FDIM)          // per-batch z elements (524288)

// 16-bf16 fragment viewed as two 16B chunks for global_load_b128 / ds_load_b128.
union FragB16 { v16bf v; uint4 q[2]; };

// ---------------------------------------------------------------- conversions
__global__ void cvt_x_kernel(const float* __restrict__ x, __bf16* __restrict__ z) {
    int i = blockIdx.x * 256 + threadIdx.x;        // 8192 blocks * 256 = 2097152 exact
    z[i] = (__bf16)x[i];
}

// WT[l][g][f] = W_l[f][g]  (bf16), so the small GEMM's A-operand rows are K-contiguous.
__global__ void cvt_w_kernel(const float* __restrict__ np, __bf16* __restrict__ wt) {
    int i = blockIdx.x * 256 + threadIdx.x;        // 32 * 256 = 8192 exact
    int l = i >> 12, r = i & 4095, g = r >> 6, f = r & 63;
    wt[(l << 12) + (g << 6) + f] = (__bf16)np[(l << 12) + (f << 6) + g];
}

// ------------------------------------------------- small GEMM: zwT = (z @ W)^T
// O[g][n] = sum_f WT[g][f] * z[n][f].  M=g(64), N=n(8192), K=f(64).
// Wave computes 64(g) x 16(n) for one batch.  grid=(128,4), block=128 (4 waves).
__global__ __launch_bounds__(128)
void gemm_zw_kernel(const __bf16* __restrict__ z, const __bf16* __restrict__ wt,
                    __bf16* __restrict__ zwT) {
    const int lane = threadIdx.x & 31;
    const int wave = threadIdx.x >> 5;
    const int col  = lane & 15, h = lane >> 4;
    const int b    = blockIdx.y;
    const int n0   = (blockIdx.x * 4 + wave) * 16;
    const __bf16* zb = z + b * BNF;

    v8f acc[4] = {};
    #pragma unroll
    for (int kk = 0; kk < 2; ++kk) {
        // B fragment: col n = n0+col, K(e) = kk*32 + 16h + e  (row-major z => contiguous)
        FragB16 bf;
        const uint4* pb = reinterpret_cast<const uint4*>(zb + (n0 + col) * FDIM + kk * 32 + 16 * h);
        bf.q[0] = pb[0]; bf.q[1] = pb[1];
        #pragma unroll
        for (int mt = 0; mt < 4; ++mt) {
            // A fragment: row g = 16*mt+col, K(e) = kk*32 + 16h + e
            FragB16 af;
            const uint4* pa = reinterpret_cast<const uint4*>(wt + (mt * 16 + col) * FDIM + kk * 32 + 16 * h);
            af.q[0] = pa[0]; af.q[1] = pa[1];
            acc[mt] = __builtin_amdgcn_wmma_f32_16x16x32_bf16(
                false, af.v, false, bf.v, (short)0, acc[mt], false, false);
        }
    }
    // C layout: reg r, lane l -> (M = r + 8h, Ncol = col).  Store transposed: zwT[g][n].
    __bf16* ob = zwT + b * BNF;
    #pragma unroll
    for (int mt = 0; mt < 4; ++mt)
        #pragma unroll
        for (int r = 0; r < 8; ++r)
            ob[(mt * 16 + r + 8 * h) * N8K + n0 + col] = (__bf16)acc[mt][r];
}

// ------------------------------------------------- big GEMM: out = relu(A @ zw)
// Block = 256 threads = 8 waves: wave w -> batch (w&3), row-half (w>>2).
// Block covers 64 A-rows x 64 cols x 4 batches.  grid = 8192/64 = 128.
// LDS: A tile 64x32 (fp32->bf16, padded rows of 40) shared by all waves (A read
// from HBM exactly once/layer) + per-batch zwT tiles 64x32 (cuts L2 re-reads).
#define ATILE_ROW 40                    // 32 + 8 pad: 16B-aligned, conflict-free b128
#define ATILE_SZ  (64 * ATILE_ROW)     // 2560 bf16
__global__ __launch_bounds__(256)
void gemm_A_kernel(const float* __restrict__ A, const __bf16* __restrict__ zwT,
                   __bf16* __restrict__ zout, float* __restrict__ fout, int final_layer) {
    __shared__ __align__(16) __bf16 smem[ATILE_SZ * 5];   // A tile + 4 B tiles (25.6 KB)
    __bf16* Atile = smem;

    const int tid  = threadIdx.x;
    const int lane = tid & 31;
    const int wave = tid >> 5;
    const int b    = wave & 3;
    const int half = wave >> 2;
    const int col  = lane & 15, h = lane >> 4;
    const int m0   = blockIdx.x * 64;

    const float4* A4 = reinterpret_cast<const float4*>(A);
    __bf16* Btile = smem + ATILE_SZ * (1 + b);

    v8f acc[2][4] = {};

    for (int it = 0; it < N8K / 32; ++it) {
        const int k0 = it * 32;
        // ---- fetch A tile (64 rows x 32 cols fp32): 512 float4 / 256 threads
        float4 av[2];
        #pragma unroll
        for (int j = 0; j < 2; ++j) {
            int idx4 = tid + 256 * j;
            int row = idx4 >> 3, c4 = (idx4 & 7) << 2;
            av[j] = A4[(m0 + row) * (N8K / 4) + ((k0 + c4) >> 2)];
        }
        // ---- fetch zwT tiles (4 batches x 64 g x 32 k bf16): 1024 uint4 / 256 threads
        uint4 bv[4];
        #pragma unroll
        for (int j = 0; j < 4; ++j) {
            int idx = tid + 256 * j;
            int bb = idx >> 8, rem = idx & 255, g = rem >> 2, qi = rem & 3;
            bv[j] = reinterpret_cast<const uint4*>(zwT + bb * BNF + g * N8K + k0)[qi];
        }
        __syncthreads();   // previous iteration's LDS reads complete
        #pragma unroll
        for (int j = 0; j < 2; ++j) {
            int idx4 = tid + 256 * j;
            int row = idx4 >> 3, c4 = (idx4 & 7) << 2;
            __bf16* d = &Atile[row * ATILE_ROW + c4];
            d[0] = (__bf16)av[j].x; d[1] = (__bf16)av[j].y;
            d[2] = (__bf16)av[j].z; d[3] = (__bf16)av[j].w;
        }
        #pragma unroll
        for (int j = 0; j < 4; ++j) {
            int idx = tid + 256 * j;
            int bb = idx >> 8, rem = idx & 255, g = rem >> 2, qi = rem & 3;
            *reinterpret_cast<uint4*>(&smem[ATILE_SZ * (1 + bb) + g * ATILE_ROW + qi * 8]) = bv[j];
        }
        __syncthreads();   // tiles visible
        // ---- B fragments from LDS: col g = 16*ft+col, K(e) = 16h + e
        FragB16 bf[4];
        #pragma unroll
        for (int ft = 0; ft < 4; ++ft) {
            const uint4* pb = reinterpret_cast<const uint4*>(&Btile[(ft * 16 + col) * ATILE_ROW + 16 * h]);
            bf[ft].q[0] = pb[0]; bf[ft].q[1] = pb[1];
        }
        // ---- A fragments from LDS + 8 WMMAs
        #pragma unroll
        for (int mt = 0; mt < 2; ++mt) {
            FragB16 af;   // row m = half*32 + mt*16 + col; K(e) = 16h + e
            const uint4* pa = reinterpret_cast<const uint4*>(
                &Atile[(half * 32 + mt * 16 + col) * ATILE_ROW + 16 * h]);
            af.q[0] = pa[0]; af.q[1] = pa[1];
            #pragma unroll
            for (int ft = 0; ft < 4; ++ft)
                acc[mt][ft] = __builtin_amdgcn_wmma_f32_16x16x32_bf16(
                    false, af.v, false, bf[ft].v, (short)0, acc[mt][ft], false, false);
        }
    }
    // ---- epilogue: fused relu; layer1 -> bf16 z (row-major), layer2 -> f32 d_out
    const int mbase = m0 + half * 32;
    if (final_layer) {
        float* ob = fout + b * BNF;
        #pragma unroll
        for (int mt = 0; mt < 2; ++mt)
            #pragma unroll
            for (int ft = 0; ft < 4; ++ft)
                #pragma unroll
                for (int r = 0; r < 8; ++r)
                    ob[(mbase + mt * 16 + r + 8 * h) * FDIM + ft * 16 + col] =
                        fmaxf(acc[mt][ft][r], 0.0f);
    } else {
        __bf16* ob = zout + b * BNF;
        #pragma unroll
        for (int mt = 0; mt < 2; ++mt)
            #pragma unroll
            for (int ft = 0; ft < 4; ++ft)
                #pragma unroll
                for (int r = 0; r < 8; ++r)
                    ob[(mbase + mt * 16 + r + 8 * h) * FDIM + ft * 16 + col] =
                        (__bf16)fmaxf(acc[mt][ft][r], 0.0f);
    }
}

// ---------------------------------------------------------------- launcher
extern "C" void kernel_launch(void* const* d_in, const int* in_sizes, int n_in,
                              void* d_out, int out_size, void* d_ws, size_t ws_size,
                              hipStream_t stream) {
    const float* x  = (const float*)d_in[0];
    // d_in[1] = t (unused)
    const float* np = (const float*)d_in[2];
    const float* A  = (const float*)d_in[3];
    float* out = (float*)d_out;

    char* ws = (char*)d_ws;
    __bf16* zbuf = (__bf16*)(ws);                      // 4 MB: z, row-major [b][n][f]
    __bf16* zw   = (__bf16*)(ws + (size_t)BATCH * BNF * 2);        // 4 MB: zwT [b][g][n]
    __bf16* wt   = (__bf16*)(ws + (size_t)2 * BATCH * BNF * 2);    // 16 KB: WT[2][64][64]

    cvt_x_kernel<<<8192, 256, 0, stream>>>(x, zbuf);
    cvt_w_kernel<<<32, 256, 0, stream>>>(np, wt);

    // layer 0
    gemm_zw_kernel<<<dim3(128, 4), 128, 0, stream>>>(zbuf, wt, zw);
    gemm_A_kernel<<<128, 256, 0, stream>>>(A, zw, zbuf, nullptr, 0);
    // layer 1
    gemm_zw_kernel<<<dim3(128, 4), 128, 0, stream>>>(zbuf, wt + 4096, zw);
    gemm_A_kernel<<<128, 256, 0, stream>>>(A, zw, nullptr, out, 1);
}
